// HA_unit_9869834846767
// MI455X (gfx1250) — compile-verified, matrix operand order checked by reading, not run on hardware
//
#include <hip/hip_runtime.h>
#include <hip/hip_bf16.h>
#include <math.h>

#define BB   4
#define ICN  256
#define CINN 512
#define NN   4096
#define EPSB 1e-5f

#if defined(__has_builtin)
#if __has_builtin(__builtin_amdgcn_global_load_async_to_lds_b128) && \
    __has_builtin(__builtin_amdgcn_s_wait_asynccnt)
#define HAVE_ASYNC_LDS 1
#endif
#endif
#ifndef HAVE_ASYNC_LDS
#define HAVE_ASYNC_LDS 0
#endif

#define AS1 __attribute__((address_space(1)))
#define AS3 __attribute__((address_space(3)))

typedef int async_v4i __attribute__((vector_size(16)));

typedef __attribute__((ext_vector_type(16))) __bf16 v16bf;
typedef __attribute__((ext_vector_type(8)))  float  v8f;
typedef __attribute__((ext_vector_type(8)))  int    v8i;

union FragU { uint4 q[2]; v16bf v; };
union FragI { uint2 q[4]; v8i v; };

__device__ __forceinline__ v8f vzero() {
  v8f z = {0.f,0.f,0.f,0.f,0.f,0.f,0.f,0.f};
  return z;
}
__device__ __forceinline__ v8i izero() {
  v8i z = {0,0,0,0,0,0,0,0};
  return z;
}

__device__ __forceinline__ v16bf frag_row_bf(const __bf16* p, int kh) {
  FragU f;
  const uint4* q = (const uint4*)p;
  f.q[0] = q[kh];
  f.q[1] = q[2 + kh];
  return f.v;
}

__device__ __forceinline__ v8i frag_row_i8(const uint8_t* p, int kh) {
  FragI f;
  const uint2* q = (const uint2*)p;
  f.q[0] = q[kh];
  f.q[1] = q[2 + kh];
  f.q[2] = q[4 + kh];
  f.q[3] = q[6 + kh];
  return f.v;
}

__device__ __forceinline__ v16bf frag_row_f32(const float* p, int kh) {
  v16bf v;
  const float* q = p + 8*kh;
#pragma unroll
  for (int j = 0; j < 8; ++j) v[j]     = (__bf16)q[j];
#pragma unroll
  for (int j = 0; j < 8; ++j) v[8 + j] = (__bf16)q[16 + j];
  return v;
}

__device__ __forceinline__ v8f wmma_bf(v16bf a, v16bf b, v8f c) {
  return __builtin_amdgcn_wmma_f32_16x16x32_bf16(false, a, false, b,
                                                 (short)0, c, false, false);
}
__device__ __forceinline__ v8i wmma_i8(v8i a, v8i b, v8i c) {
  return __builtin_amdgcn_wmma_i32_16x16x64_iu8(false, a, false, b, c,
                                                false, false);
}

__global__ void k_conv1(const float* __restrict__ x,
                        const float* __restrict__ w1, const float* __restrict__ w1b,
                        const float* __restrict__ g,  const float* __restrict__ be,
                        const float* __restrict__ mu, const float* __restrict__ va,
                        const float* __restrict__ nw, const float* __restrict__ nb,
                        __bf16* __restrict__ wfeat, __bf16* __restrict__ nodes) {
  __shared__ float xs[32][17];
  int lane = threadIdx.x;
  int n0 = blockIdx.x * 16, o0 = blockIdx.y * 64, b = blockIdx.z;
  int nn = lane & 15, kh = lane >> 4;
  v8f accW[4], accN[4];
#pragma unroll
  for (int i = 0; i < 4; ++i) { accW[i] = vzero(); accN[i] = vzero(); }
  const float* xb = x + (size_t)b * CINN * NN;
  for (int c0 = 0; c0 < CINN; c0 += 32) {
    const float* xr = xb + (size_t)(c0 + lane) * NN + n0;
#pragma unroll
    for (int j = 0; j < 16; ++j) xs[lane][j] = xr[j];
    __syncthreads();
    v16bf a;
#pragma unroll
    for (int j = 0; j < 8; ++j) a[j]     = (__bf16)xs[8*kh + j][nn];
#pragma unroll
    for (int j = 0; j < 8; ++j) a[8 + j] = (__bf16)xs[8*kh + 16 + j][nn];
#pragma unroll
    for (int t = 0; t < 4; ++t) {
      int o = o0 + t*16 + nn;
      accW[t] = wmma_bf(a, frag_row_f32(w1 + (size_t)o*CINN + c0, kh), accW[t]);
      accN[t] = wmma_bf(a, frag_row_f32(nw + (size_t)o*CINN + c0, kh), accN[t]);
    }
    __syncthreads();
  }
#pragma unroll
  for (int t = 0; t < 4; ++t) {
    int o = o0 + t*16 + nn;
    float inv = g[o] * rsqrtf(va[o] + EPSB);
    float bi  = be[o] - mu[o]*inv + w1b[o]*inv;
    float nbo = nb[o];
#pragma unroll
    for (int r = 0; r < 8; ++r) {
      int n = n0 + r + 8*kh;
      wfeat[((size_t)b*NN + n)*ICN + o] = (__bf16)(accW[t][r]*inv + bi);
      nodes[((size_t)b*ICN + o)*NN + n] = (__bf16)(accN[t][r] + nbo);
    }
  }
}

__global__ void k_cat(const float* __restrict__ x, __bf16* __restrict__ cat) {
  __shared__ float ts[16][17];
  int t = threadIdx.x, r = t >> 4, q = t & 15;
  int n0 = blockIdx.x * 16, c0 = blockIdx.y * 16, b = blockIdx.z;
  ts[r][q] = x[((size_t)b*CINN + c0 + r)*NN + n0 + q];
  __syncthreads();
  cat[((size_t)b*NN + n0 + r)*(2*ICN) + c0 + q] = (__bf16)ts[q][r];
}

__global__ void k_gram(const __bf16* __restrict__ wfeat, __bf16* __restrict__ iw) {
  int lane = threadIdx.x;
  int n0 = blockIdx.x * 16, m0 = blockIdx.y * 64, b = blockIdx.z;
  int nn = lane & 15, kh = lane >> 4;
  v8f acc[4];
#pragma unroll
  for (int i = 0; i < 4; ++i) acc[i] = vzero();
  const __bf16* wb = wfeat + (size_t)b * NN * ICN;
  for (int k0 = 0; k0 < ICN; k0 += 32) {
    v16bf a = frag_row_bf(wb + (size_t)(n0 + nn)*ICN + k0, kh);
#pragma unroll
    for (int t = 0; t < 4; ++t)
      acc[t] = wmma_bf(a, frag_row_bf(wb + (size_t)(m0 + t*16 + nn)*ICN + k0, kh), acc[t]);
  }
#pragma unroll
  for (int t = 0; t < 4; ++t) {
    int m = m0 + t*16 + nn;
#pragma unroll
    for (int r = 0; r < 8; ++r) {
      int n = n0 + r + 8*kh;
      iw[((size_t)b*NN + n)*NN + m] = (__bf16)(acc[t][r] * 0.0625f);
    }
  }
}

__global__ void k_bin(const __bf16* __restrict__ iw, const float* __restrict__ delta,
                      __bf16* __restrict__ b0bf, uint8_t* __restrict__ b0i8) {
  float d = delta[0];
  float thr = logf(d / (1.0f - d));
  size_t i = ((size_t)blockIdx.x * blockDim.x + threadIdx.x) * 8;
  union { uint4 u; __bf16 h[8]; } in, ob;
  in.u = *(const uint4*)(iw + i);
  uint32_t lo = 0, hi = 0;
#pragma unroll
  for (int j = 0; j < 8; ++j) {
    bool bit = ((float)in.h[j] >= thr);
    ob.h[j] = (__bf16)(bit ? 1.0f : 0.0f);
    uint32_t by = bit ? 1u : 0u;
    if (j < 4) lo |= by << (8*j); else hi |= by << (8*(j-4));
  }
  *(uint4*)(b0bf + i) = ob.u;
  uint2 pk; pk.x = lo; pk.y = hi;
  *(uint2*)(b0i8 + i) = pk;
}

__global__ void k_pow2(const uint8_t* __restrict__ b0, __bf16* __restrict__ P2) {
  __shared__ __align__(16) uint8_t bs[2][64 * 144];
  int tid = threadIdx.x;
  int lane = tid & 31, w = tid >> 5;
  int n0 = blockIdx.x * 128 + w * 32, m0 = blockIdx.y * 64, b = blockIdx.z;
  int nn = lane & 15, kh = lane >> 4;
  v8i acc[2][4];
#pragma unroll
  for (int s = 0; s < 2; ++s)
#pragma unroll
    for (int t = 0; t < 4; ++t) acc[s][t] = izero();
  const uint8_t* bb = b0 + (size_t)b * NN * NN;

  auto stage = [&](int buf, int kp) {
#if HAVE_ASYNC_LDS
#pragma unroll
    for (int j = 0; j < 4; ++j) {
      int c = tid + 128*j;
      int row = c >> 3, col = (c & 7) * 16;
      const uint8_t* src = bb + (size_t)(m0 + row)*NN + kp + col;
      uint8_t* dst = &bs[buf][row*144 + col];
      __builtin_amdgcn_global_load_async_to_lds_b128((AS1 async_v4i*)src,
                                                     (AS3 async_v4i*)dst, 0, 0);
    }
#else
    int row = tid >> 1, off = (tid & 1) * 64;
    const uint4* src = (const uint4*)(bb + (size_t)(m0 + row)*NN + kp + off);
    uint4* dst = (uint4*)(&bs[buf][row*144 + off]);
#pragma unroll
    for (int j = 0; j < 4; ++j) dst[j] = src[j];
#endif
  };

  stage(0, 0);
#if HAVE_ASYNC_LDS
  __builtin_amdgcn_s_wait_asynccnt(0);
#endif
  __syncthreads();
  const int npan = NN / 128;
  for (int p = 0; p < npan; ++p) {
    int kp = p * 128, cur = p & 1;
    if (p + 1 < npan) stage(1 - cur, kp + 128);
#pragma unroll
    for (int ks = 0; ks < 2; ++ks) {
      int k0 = kp + ks*64;
      v8i a0 = frag_row_i8(bb + (size_t)(n0 + nn)*NN + k0, kh);
      v8i a1 = frag_row_i8(bb + (size_t)(n0 + 16 + nn)*NN + k0, kh);
#pragma unroll
      for (int t = 0; t < 4; ++t) {
        v8i bf = frag_row_i8(&bs[cur][(t*16 + nn)*144 + ks*64], kh);
        acc[0][t] = wmma_i8(a0, bf, acc[0][t]);
        acc[1][t] = wmma_i8(a1, bf, acc[1][t]);
      }
    }
#if HAVE_ASYNC_LDS
    __builtin_amdgcn_s_wait_asynccnt(0);
#endif
    __syncthreads();
  }
#pragma unroll
  for (int s = 0; s < 2; ++s)
#pragma unroll
    for (int t = 0; t < 4; ++t) {
      int m = m0 + t*16 + nn;
#pragma unroll
      for (int r = 0; r < 8; ++r) {
        int n = n0 + s*16 + r + 8*kh;
        P2[((size_t)b*NN + n)*NN + m] = (__bf16)(float)acc[s][t][r];
      }
    }
}

__global__ void k_pow3(const __bf16* __restrict__ P2, const __bf16* __restrict__ b0bf,
                       __bf16* __restrict__ P3) {
  __shared__ __align__(16) uint8_t bs[2][64 * 272];
  int tid = threadIdx.x;
  int lane = tid & 31, w = tid >> 5;
  int n0 = blockIdx.x * 128 + w * 32, m0 = blockIdx.y * 64, b = blockIdx.z;
  int nn = lane & 15, kh = lane >> 4;
  v8f acc[2][4];
#pragma unroll
  for (int s = 0; s < 2; ++s)
#pragma unroll
    for (int t = 0; t < 4; ++t) acc[s][t] = vzero();
  const __bf16* ab = P2   + (size_t)b * NN * NN;
  const __bf16* bb = b0bf + (size_t)b * NN * NN;
  const uint8_t* bby = (const uint8_t*)bb;

  auto stage = [&](int buf, int kp) {
#if HAVE_ASYNC_LDS
#pragma unroll
    for (int j = 0; j < 8; ++j) {
      int c = tid + 128*j;
      int row = c >> 4, col = (c & 15) * 16;
      const uint8_t* src = bby + (size_t)(m0 + row)*(NN*2) + kp*2 + col;
      uint8_t* dst = &bs[buf][row*272 + col];
      __builtin_amdgcn_global_load_async_to_lds_b128((AS1 async_v4i*)src,
                                                     (AS3 async_v4i*)dst, 0, 0);
    }
#else
    int row = tid >> 1, off = (tid & 1) * 128;
    const uint4* src = (const uint4*)(bby + (size_t)(m0 + row)*(NN*2) + kp*2 + off);
    uint4* dst = (uint4*)(&bs[buf][row*272 + off]);
#pragma unroll
    for (int j = 0; j < 8; ++j) dst[j] = src[j];
#endif
  };

  stage(0, 0);
#if HAVE_ASYNC_LDS
  __builtin_amdgcn_s_wait_asynccnt(0);
#endif
  __syncthreads();
  const int npan = NN / 128;
  for (int p = 0; p < npan; ++p) {
    int kp = p * 128, cur = p & 1;
    if (p + 1 < npan) stage(1 - cur, kp + 128);
#pragma unroll
    for (int ks = 0; ks < 4; ++ks) {
      int k0 = kp + ks*32;
      v16bf a0 = frag_row_bf(ab + (size_t)(n0 + nn)*NN + k0, kh);
      v16bf a1 = frag_row_bf(ab + (size_t)(n0 + 16 + nn)*NN + k0, kh);
#pragma unroll
      for (int t = 0; t < 4; ++t) {
        v16bf bf = frag_row_bf((const __bf16*)(&bs[cur][(t*16 + nn)*272 + ks*64]), kh);
        acc[0][t] = wmma_bf(a0, bf, acc[0][t]);
        acc[1][t] = wmma_bf(a1, bf, acc[1][t]);
      }
    }
#if HAVE_ASYNC_LDS
    __builtin_amdgcn_s_wait_asynccnt(0);
#endif
    __syncthreads();
  }
#pragma unroll
  for (int s = 0; s < 2; ++s)
#pragma unroll
    for (int t = 0; t < 4; ++t) {
      int m = m0 + t*16 + nn;
#pragma unroll
      for (int r = 0; r < 8; ++r) {
        int n = n0 + s*16 + r + 8*kh;
        P3[((size_t)b*NN + n)*NN + m] = (__bf16)acc[s][t][r];
      }
    }
}

__global__ void k_soft(const __bf16* __restrict__ iw, const __bf16* __restrict__ P,
                       float* __restrict__ rowmax, float* __restrict__ rowrcp) {
  int lane = threadIdx.x;
  int n = blockIdx.x, b = blockIdx.y;
  const __bf16* row = iw + ((size_t)b*NN + n)*NN;
  const __bf16* pr  = P  + ((size_t)b*NN + n)*NN;
  float m = -3.0e38f;
  for (int k = lane; k < NN; k += 32)
    m = fmaxf(m, (float)pr[k] * (float)row[k]);
#pragma unroll
  for (int off = 16; off; off >>= 1) m = fmaxf(m, __shfl_xor(m, off, 32));
  float s = 0.f;
  for (int k = lane; k < NN; k += 32)
    s += __expf((float)pr[k] * (float)row[k] - m);
#pragma unroll
  for (int off = 16; off; off >>= 1) s += __shfl_xor(s, off, 32);
  if (lane == 0) {
    rowmax[(size_t)b*NN + n] = m;
    rowrcp[(size_t)b*NN + n] = 1.0f / s;
  }
}

__global__ void k_agg(const __bf16* __restrict__ iw, const __bf16* __restrict__ P,
                      const __bf16* __restrict__ nodes,
                      const float* __restrict__ rowmax, const float* __restrict__ rowrcp,
                      __bf16* __restrict__ tout) {
  int lane = threadIdx.x;
  int n0 = blockIdx.x * 16, c0 = blockIdx.y * 128, b = blockIdx.z;
  int nn = lane & 15, kh = lane >> 4;
  int n = n0 + nn;
  float mx = rowmax[(size_t)b*NN + n];
  float rc = rowrcp[(size_t)b*NN + n];
  const __bf16* irow = iw + ((size_t)b*NN + n)*NN;
  const __bf16* prow = P  + ((size_t)b*NN + n)*NN;
  const __bf16* nbse = nodes + (size_t)b * ICN * NN;
  v8f acc[8];
#pragma unroll
  for (int i = 0; i < 8; ++i) acc[i] = vzero();
  for (int k0 = 0; k0 < NN; k0 += 32) {
    __builtin_prefetch(irow + k0 + 512, 0, 3);
    __builtin_prefetch(prow + k0 + 512, 0, 3);
    v16bf fi = frag_row_bf(irow + k0, kh);
    v16bf fp = frag_row_bf(prow + k0, kh);
    v16bf a;
#pragma unroll
    for (int j = 0; j < 16; ++j) {
      float v = (float)fp[j] * (float)fi[j];
      a[j] = (__bf16)(__expf(v - mx) * rc);
    }
#pragma unroll
    for (int t = 0; t < 8; ++t)
      acc[t] = wmma_bf(a, frag_row_bf(nbse + (size_t)(c0 + t*16 + nn)*NN + k0, kh), acc[t]);
  }
#pragma unroll
  for (int t = 0; t < 8; ++t) {
    int c = c0 + t*16 + nn;
#pragma unroll
    for (int r = 0; r < 8; ++r) {
      int nr = n0 + r + 8*kh;
      tout[((size_t)b*NN + nr)*ICN + c] = (__bf16)acc[t][r];
    }
  }
}

__global__ void k_hop(const __bf16* __restrict__ tin, const float* __restrict__ hw,
                      const float* __restrict__ hb, __bf16* __restrict__ xp, int hop) {
  int lane = threadIdx.x;
  int n0 = blockIdx.x * 16, o0 = blockIdx.y * 64, b = blockIdx.z;
  int nn = lane & 15, kh = lane >> 4;
  v8f acc[4];
#pragma unroll
  for (int i = 0; i < 4; ++i) acc[i] = vzero();
  const __bf16* trow = tin + ((size_t)b*NN + n0 + nn)*ICN;
  for (int k0 = 0; k0 < ICN; k0 += 32) {
    v16bf a = frag_row_bf(trow + k0, kh);
#pragma unroll
    for (int t = 0; t < 4; ++t)
      acc[t] = wmma_bf(a, frag_row_f32(hw + (size_t)(o0 + t*16 + nn)*ICN + k0, kh), acc[t]);
  }
#pragma unroll
  for (int t = 0; t < 4; ++t) {
    int o = o0 + t*16 + nn;
    float bo = hb[o];
#pragma unroll
    for (int r = 0; r < 8; ++r) {
      int n = n0 + r + 8*kh;
      xp[((size_t)b*NN + n)*(3*ICN) + hop*ICN + o] = (__bf16)(acc[t][r] + bo);
    }
  }
}

__global__ void k_fuse(const __bf16* __restrict__ xp, const float* __restrict__ fw,
                       const float* __restrict__ fb,
                       const float* __restrict__ g,  const float* __restrict__ be,
                       const float* __restrict__ mu, const float* __restrict__ va,
                       __bf16* __restrict__ cat) {
  int lane = threadIdx.x;
  int n0 = blockIdx.x * 16, o0 = blockIdx.y * 64, b = blockIdx.z;
  int nn = lane & 15, kh = lane >> 4;
  v8f acc[4];
#pragma unroll
  for (int i = 0; i < 4; ++i) acc[i] = vzero();
  const __bf16* xrow = xp + ((size_t)b*NN + n0 + nn)*(3*ICN);
  for (int k0 = 0; k0 < 3*ICN; k0 += 32) {
    v16bf a = frag_row_bf(xrow + k0, kh);
#pragma unroll
    for (int t = 0; t < 4; ++t)
      acc[t] = wmma_bf(a, frag_row_f32(fw + (size_t)(o0 + t*16 + nn)*(3*ICN) + k0, kh), acc[t]);
  }
#pragma unroll
  for (int t = 0; t < 4; ++t) {
    int o = o0 + t*16 + nn;
    float inv = g[o] * rsqrtf(va[o] + EPSB);
    float bi  = be[o] - mu[o]*inv + fb[o]*inv;
#pragma unroll
    for (int r = 0; r < 8; ++r) {
      int n = n0 + r + 8*kh;
      cat[((size_t)b*NN + n)*(2*ICN) + ICN + o] = (__bf16)(acc[t][r]*inv + bi);
    }
  }
}

__global__ void k_res(const __bf16* __restrict__ cat, const float* __restrict__ rw,
                      const float* __restrict__ rb,
                      const float* __restrict__ g,  const float* __restrict__ be,
                      const float* __restrict__ mu, const float* __restrict__ va,
                      float* __restrict__ out) {
  int lane = threadIdx.x;
  int o0 = blockIdx.x * 16, n0 = blockIdx.y * 64, b = blockIdx.z;
  int nn = lane & 15, kh = lane >> 4;
  v8f acc[4];
#pragma unroll
  for (int i = 0; i < 4; ++i) acc[i] = vzero();
  const float* arow = rw + (size_t)(o0 + nn)*CINN;
  for (int k0 = 0; k0 < CINN; k0 += 32) {
    v16bf a = frag_row_f32(arow + k0, kh);
#pragma unroll
    for (int t = 0; t < 4; ++t)
      acc[t] = wmma_bf(a, frag_row_bf(cat + ((size_t)b*NN + n0 + t*16 + nn)*CINN + k0, kh), acc[t]);
  }
#pragma unroll
  for (int t = 0; t < 4; ++t) {
    int n = n0 + t*16 + nn;
#pragma unroll
    for (int r = 0; r < 8; ++r) {
      int o = o0 + r + 8*kh;
      float inv = g[o] * rsqrtf(va[o] + EPSB);
      float bi  = be[o] - mu[o]*inv + rb[o]*inv;
      out[((size_t)b*CINN + o)*NN + n] = acc[t][r]*inv + bi;
    }
  }
}

extern "C" void kernel_launch(void* const* d_in, const int* in_sizes, int n_in,
                              void* d_out, int out_size, void* d_ws, size_t ws_size,
                              hipStream_t stream) {
  (void)in_sizes; (void)n_in; (void)out_size; (void)ws_size;
  const float* x     = (const float*)d_in[0];
  const float* delta = (const float*)d_in[1];
  const float* w1_w  = (const float*)d_in[2];
  const float* w1_b  = (const float*)d_in[3];
  const float* bn1_g = (const float*)d_in[4];
  const float* bn1_b = (const float*)d_in[5];
  const float* bn1_m = (const float*)d_in[6];
  const float* bn1_v = (const float*)d_in[7];
  const float* nodew = (const float*)d_in[8];
  const float* nodeb = (const float*)d_in[9];
  const float* hop_w = (const float*)d_in[10];
  const float* hop_b = (const float*)d_in[11];
  const float* fusew = (const float*)d_in[12];
  const float* fuseb = (const float*)d_in[13];
  const float* bnf_g = (const float*)d_in[14];
  const float* bnf_b = (const float*)d_in[15];
  const float* bnf_m = (const float*)d_in[16];
  const float* bnf_v = (const float*)d_in[17];
  const float* res_w = (const float*)d_in[18];
  const float* res_b = (const float*)d_in[19];
  const float* bnr_g = (const float*)d_in[20];
  const float* bnr_b = (const float*)d_in[21];
  const float* bnr_m = (const float*)d_in[22];
  const float* bnr_v = (const float*)d_in[23];

  uint8_t* ws = (uint8_t*)d_ws;
  size_t off = 0;
  __bf16* wfeat = (__bf16*)(ws + off); off += (size_t)BB*NN*ICN*2;
  __bf16* nodes = (__bf16*)(ws + off); off += (size_t)BB*NN*ICN*2;
  __bf16* tbuf  = (__bf16*)(ws + off); off += (size_t)BB*NN*ICN*2;
  __bf16* xp    = (__bf16*)(ws + off); off += (size_t)BB*NN*(3*ICN)*2;
  __bf16* cat   = (__bf16*)(ws + off); off += (size_t)BB*NN*(2*ICN)*2;
  float*  rmax  = (float*)(ws + off);  off += (size_t)BB*NN*4;
  float*  rrcp  = (float*)(ws + off);  off += (size_t)BB*NN*4;
  __bf16* iw    = (__bf16*)(ws + off); off += (size_t)BB*NN*NN*2;
  __bf16* b0bf  = (__bf16*)(ws + off); off += (size_t)BB*NN*NN*2;
  __bf16* P2    = (__bf16*)(ws + off); off += (size_t)BB*NN*NN*2;
  __bf16* P3    = (__bf16*)(ws + off); off += (size_t)BB*NN*NN*2;
  uint8_t* b0i8 = (uint8_t*)(ws + off); off += (size_t)BB*NN*NN;

  dim3 w1b(32);
  k_conv1<<<dim3(NN/16, ICN/64, BB), w1b, 0, stream>>>(
      x, w1_w, w1_b, bn1_g, bn1_b, bn1_m, bn1_v, nodew, nodeb, wfeat, nodes);
  k_cat<<<dim3(NN/16, ICN/16, BB), dim3(256), 0, stream>>>(x, cat);
  k_gram<<<dim3(NN/16, NN/64, BB), w1b, 0, stream>>>(wfeat, iw);
  {
    size_t total = (size_t)BB * NN * NN;
    unsigned blocks = (unsigned)(total / (8ull * 256ull));
    k_bin<<<dim3(blocks), dim3(256), 0, stream>>>(iw, delta, b0bf, b0i8);
  }
  k_pow2<<<dim3(NN/128, NN/64, BB), dim3(128), 0, stream>>>(b0i8, P2);
  k_pow3<<<dim3(NN/128, NN/64, BB), dim3(128), 0, stream>>>(P2, b0bf, P3);

  const __bf16* Ps[3] = {b0bf, P2, P3};
  for (int hop = 0; hop < 3; ++hop) {
    k_soft<<<dim3(NN, BB), w1b, 0, stream>>>(iw, Ps[hop], rmax, rrcp);
    k_agg<<<dim3(NN/16, ICN/128, BB), w1b, 0, stream>>>(
        iw, Ps[hop], nodes, rmax, rrcp, tbuf);
    k_hop<<<dim3(NN/16, ICN/64, BB), w1b, 0, stream>>>(
        tbuf, hop_w + (size_t)hop*ICN*ICN, hop_b + (size_t)hop*ICN, xp, hop);
  }
  k_fuse<<<dim3(NN/16, ICN/64, BB), w1b, 0, stream>>>(
      xp, fusew, fuseb, bnf_g, bnf_b, bnf_m, bnf_v, cat);
  k_res<<<dim3(CINN/16, NN/64, BB), w1b, 0, stream>>>(
      cat, res_w, res_b, bnr_g, bnr_b, bnr_m, bnr_v, (float*)d_out);
}
